// KimiLayer_39874476376651
// MI455X (gfx1250) — compile-verified, matrix operand order checked by reading, not run on hardware
//
#include <hip/hip_runtime.h>
#include <stdint.h>

// ---------------------------------------------------------------------------
// Types / helpers for CDNA5 WMMA (gfx1250, wave32)
// ---------------------------------------------------------------------------
typedef __attribute__((ext_vector_type(16))) __bf16 v16bf;
typedef __attribute__((ext_vector_type(8)))  float  v8f;

static __device__ __forceinline__ __bf16 f2bf(float f) { return (__bf16)f; }
static __device__ __forceinline__ float bf2f(__bf16 h) { return (float)h; }

// Async global -> LDS copy (gfx1250). VDST holds the wave-relative LDS byte
// address (low 32 bits of a flat shared pointer), VADDR the 64-bit global
// address. Tracked by ASYNCcnt.
static __device__ __forceinline__ void async_copy_b128(const void* gptr,
                                                       void* lptr) {
  unsigned lds = (unsigned)(size_t)lptr;
  asm volatile("global_load_async_to_lds_b128 %0, %1, off"
               :: "v"(lds), "v"(gptr) : "memory");
}
static __device__ __forceinline__ void async_wait0() {
  asm volatile("s_wait_asynccnt 0" ::: "memory");
}

// Load one 16x32 bf16 A/B fragment from a row-major [rows][strideHalves]
// bf16 tile. ISA 16-bit layout: lane L -> matrix row (L&15), K-base
// ((L>>4)*8); halves 0..7 cover K = kb..kb+7, halves 8..15 cover K+16.
static __device__ __forceinline__ v16bf load_frag(const __bf16* base,
                                                  int strideHalves,
                                                  int row0, int col0) {
  const int lane = threadIdx.x & 31;
  const __bf16* p = base + (size_t)(row0 + (lane & 15)) * strideHalves
                         + col0 + ((lane >> 4) << 3);
  union { v16bf v; uint4 q[2]; } u;
  u.q[0] = *(const uint4*)p;
  u.q[1] = *(const uint4*)(p + 16);
  return u.v;
}

static __device__ __forceinline__ v8f wmma_bf16(v16bf a, v16bf b, v8f c) {
  return __builtin_amdgcn_wmma_f32_16x16x32_bf16(
      /*neg_a=*/false, a, /*neg_b=*/false, b,
      /*c_mod=*/(short)0, c, /*reuse_a=*/false, /*reuse_b=*/false);
}

// Convert 16 f32 (4x float4) -> 16 bf16 and store 32B to LDS.
static __device__ __forceinline__ void cvt_store_w(const float4* f,
                                                   __bf16* dst, bool valid) {
  union { uint4 q[2]; __bf16 h[16]; } t;
  if (valid) {
#pragma unroll
    for (int j = 0; j < 4; ++j) {
      float4 v = f[j];
      t.h[j * 4 + 0] = f2bf(v.x); t.h[j * 4 + 1] = f2bf(v.y);
      t.h[j * 4 + 2] = f2bf(v.z); t.h[j * 4 + 3] = f2bf(v.w);
    }
  } else {
#pragma unroll
    for (int j = 0; j < 16; ++j) t.h[j] = f2bf(0.f);
  }
  *(uint4*)dst = t.q[0];
  *(uint4*)(dst + 8) = t.q[1];
}

// ---------------------------------------------------------------------------
// Problem constants
// ---------------------------------------------------------------------------
#define HDIM   2048
#define TOKENS 2048   // B*S
#define NHEAD  16
#define QHD    192
#define NOPE   128
#define VDIM   128
#define NEXP   16
#define IDIM   1408
#define SHIDIM 2816

// ---------------------------------------------------------------------------
// RMSNorm: f32 [T,H] -> bf16 [T,H]
// ---------------------------------------------------------------------------
__global__ __launch_bounds__(256) void rmsnorm_kernel(
    const float* __restrict__ x, const float* __restrict__ g,
    __bf16* __restrict__ out, int H) {
  const int t = blockIdx.x;
  const float* row = x + (size_t)t * H;
  float ss = 0.f;
  for (int i = threadIdx.x; i < H; i += 256) { float v = row[i]; ss += v * v; }
#pragma unroll
  for (int off = 16; off >= 1; off >>= 1) ss += __shfl_xor(ss, off, 32);
  __shared__ float part[8];
  __shared__ float stot;
  const int lane = threadIdx.x & 31, wid = threadIdx.x >> 5;
  if (lane == 0) part[wid] = ss;
  __syncthreads();
  if (threadIdx.x == 0) {
    float s = 0.f;
    for (int i = 0; i < 8; ++i) s += part[i];
    stot = rsqrtf(s / (float)H + 1e-6f);
  }
  __syncthreads();
  const float r = stot;
  for (int i = threadIdx.x; i < H; i += 256)
    out[(size_t)t * H + i] = f2bf(row[i] * r * g[i]);
}

// ---------------------------------------------------------------------------
// Dense GEMM: C[M,N] = A_bf16[M,K] * W_f32[N,K]^T  (+residual), out f32/bf16.
// Block = 256 thr (8 waves), tile 128x128, k-step 32, wave = 64x32.
// Double-buffered LDS; A tiles via async global->LDS copies.
// ---------------------------------------------------------------------------
__global__ __launch_bounds__(256, 1) void gemm_dense_kernel(
    const __bf16* __restrict__ A, int lda,
    const float* __restrict__ W, int ldw,
    float* __restrict__ outF, __bf16* __restrict__ outB,
    const float* __restrict__ residual,
    int M, int N, int K) {
  __shared__ __bf16 As[2][128 * 40];
  __shared__ __bf16 Ws[2][128 * 40];
  const int tid = threadIdx.x;
  const int lane = tid & 31;
  const int wid = tid >> 5;
  const int wm = wid >> 2;   // 0..1
  const int wn = wid & 3;    // 0..3
  const int m0 = blockIdx.x * 128;
  const int n0 = blockIdx.y * 128;
  const int lr = tid >> 1;          // tile row 0..127
  const int lc = (tid & 1) * 16;    // 0 or 16 elements

  // Clamp OOB rows to a valid address: garbage feeds only C rows never stored.
  const int aRow = (m0 + lr < M) ? (m0 + lr) : 0;
  const __bf16* aBase = A + (size_t)aRow * lda + lc;
  const bool wValid = (n0 + lr) < N;
  const float* wBase = W + (size_t)(wValid ? (n0 + lr) : 0) * ldw + lc;
  __bf16* aDst0 = &As[0][lr * 40 + lc];
  __bf16* aDst1 = &As[1][lr * 40 + lc];
  __bf16* wDst0 = &Ws[0][lr * 40 + lc];
  __bf16* wDst1 = &Ws[1][lr * 40 + lc];

  v8f c[4][2];
#pragma unroll
  for (int mi = 0; mi < 4; ++mi)
#pragma unroll
    for (int ni = 0; ni < 2; ++ni)
#pragma unroll
      for (int r = 0; r < 8; ++r) c[mi][ni][r] = 0.f;

  const int nk = K >> 5;
  float4 wreg[4];

  // prologue: stage k-step 0 into buffer 0
  async_copy_b128(aBase, aDst0);
  async_copy_b128(aBase + 8, aDst0 + 8);
  {
    const float4* s = (const float4*)wBase;
    wreg[0] = s[0]; wreg[1] = s[1]; wreg[2] = s[2]; wreg[3] = s[3];
  }
  cvt_store_w(wreg, wDst0, wValid);
  async_wait0();
  __syncthreads();

  for (int kk = 0; kk < nk; ++kk) {
    const int cur = kk & 1;
    const bool more = (kk + 1) < nk;
    if (more) {  // issue next tile's global traffic before this tile's math
      const __bf16* an = aBase + (size_t)(kk + 1) * 32;
      async_copy_b128(an, cur ? aDst0 : aDst1);
      async_copy_b128(an + 8, (cur ? aDst0 : aDst1) + 8);
      const float4* s = (const float4*)(wBase + (size_t)(kk + 1) * 32);
      wreg[0] = s[0]; wreg[1] = s[1]; wreg[2] = s[2]; wreg[3] = s[3];
    }
    v16bf a[4];
#pragma unroll
    for (int mi = 0; mi < 4; ++mi)
      a[mi] = load_frag(As[cur], 40, wm * 64 + mi * 16, 0);
#pragma unroll
    for (int ni = 0; ni < 2; ++ni) {
      v16bf b = load_frag(Ws[cur], 40, wn * 32 + ni * 16, 0);
#pragma unroll
      for (int mi = 0; mi < 4; ++mi)
        c[mi][ni] = wmma_bf16(a[mi], b, c[mi][ni]);
    }
    if (more) {
      cvt_store_w(wreg, cur ? wDst0 : wDst1, wValid);
      async_wait0();
    }
    __syncthreads();
  }

  const int mo = (lane >> 4) << 3;
  const int nn = lane & 15;
#pragma unroll
  for (int mi = 0; mi < 4; ++mi)
#pragma unroll
    for (int ni = 0; ni < 2; ++ni)
#pragma unroll
      for (int r = 0; r < 8; ++r) {
        const int m = m0 + wm * 64 + mi * 16 + r + mo;
        const int n = n0 + wn * 32 + ni * 16 + nn;
        if (m < M && n < N) {
          float v = c[mi][ni][r];
          if (residual) v += residual[(size_t)m * N + n];
          if (outF) outF[(size_t)m * N + n] = v;
          if (outB) outB[(size_t)m * N + n] = f2bf(v);
        }
      }
}

// ---------------------------------------------------------------------------
// Dual gate/up GEMM + fused SiLU.
//   act[row, n] = silu(x2 * Wg^T) * (x2 * Wu^T),  Wg = W + e*strideE,
//   Wu = Wg + Ih*K.  Optional token gather (grouped-MoE) via counts/tok_list.
// ---------------------------------------------------------------------------
__global__ __launch_bounds__(256, 1) void gemm_gu_silu_kernel(
    const __bf16* __restrict__ A, int lda,
    const float* __restrict__ Wbase, long long wStrideE, int Ih, int K,
    const int* __restrict__ counts, const int* __restrict__ tok_list,
    int Mdense, __bf16* __restrict__ act) {
  const int e = blockIdx.z;
  const int cnt = counts ? counts[e] : Mdense;
  const int m0 = blockIdx.x * 128;
  if (m0 >= cnt) return;
  const int n0 = blockIdx.y * 128;
  const float* Wg = Wbase + (size_t)e * (size_t)wStrideE;
  const float* Wu = Wg + (size_t)Ih * K;
  const int* tl = tok_list ? (tok_list + e * TOKENS) : (const int*)0;
  const int rowBase = counts ? e * TOKENS : 0;

  __shared__ __bf16 As[2][128 * 40];
  __shared__ __bf16 Gs[2][128 * 40];
  __shared__ __bf16 Us[2][128 * 40];
  const int tid = threadIdx.x;
  const int lane = tid & 31;
  const int wid = tid >> 5;
  const int wm = wid >> 2, wn = wid & 3;
  const int lr = tid >> 1;
  const int lc = (tid & 1) * 16;

  const int mg = m0 + lr;
  const int aRow = (mg < cnt) ? (tl ? tl[mg] : mg) : 0;
  const __bf16* aBase = A + (size_t)aRow * lda + lc;
  const int ng = n0 + lr;   // Ih is a multiple of 128: always in range
  const float* gBase = Wg + (size_t)ng * K + lc;
  const float* uBase = Wu + (size_t)ng * K + lc;
  __bf16* aDst0 = &As[0][lr * 40 + lc];
  __bf16* aDst1 = &As[1][lr * 40 + lc];
  __bf16* gDst0 = &Gs[0][lr * 40 + lc];
  __bf16* gDst1 = &Gs[1][lr * 40 + lc];
  __bf16* uDst0 = &Us[0][lr * 40 + lc];
  __bf16* uDst1 = &Us[1][lr * 40 + lc];

  v8f cg[4][2], cu[4][2];
#pragma unroll
  for (int mi = 0; mi < 4; ++mi)
#pragma unroll
    for (int ni = 0; ni < 2; ++ni)
#pragma unroll
      for (int r = 0; r < 8; ++r) { cg[mi][ni][r] = 0.f; cu[mi][ni][r] = 0.f; }

  const int nk = K >> 5;
  float4 greg[4], ureg[4];

  async_copy_b128(aBase, aDst0);
  async_copy_b128(aBase + 8, aDst0 + 8);
  {
    const float4* sg = (const float4*)gBase;
    const float4* su = (const float4*)uBase;
#pragma unroll
    for (int j = 0; j < 4; ++j) { greg[j] = sg[j]; ureg[j] = su[j]; }
  }
  cvt_store_w(greg, gDst0, true);
  cvt_store_w(ureg, uDst0, true);
  async_wait0();
  __syncthreads();

  for (int kk = 0; kk < nk; ++kk) {
    const int cur = kk & 1;
    const bool more = (kk + 1) < nk;
    if (more) {
      const __bf16* an = aBase + (size_t)(kk + 1) * 32;
      async_copy_b128(an, cur ? aDst0 : aDst1);
      async_copy_b128(an + 8, (cur ? aDst0 : aDst1) + 8);
      const float4* sg = (const float4*)(gBase + (size_t)(kk + 1) * 32);
      const float4* su = (const float4*)(uBase + (size_t)(kk + 1) * 32);
#pragma unroll
      for (int j = 0; j < 4; ++j) { greg[j] = sg[j]; ureg[j] = su[j]; }
    }
    v16bf a[4];
#pragma unroll
    for (int mi = 0; mi < 4; ++mi)
      a[mi] = load_frag(As[cur], 40, wm * 64 + mi * 16, 0);
#pragma unroll
    for (int ni = 0; ni < 2; ++ni) {
      v16bf bg = load_frag(Gs[cur], 40, wn * 32 + ni * 16, 0);
      v16bf bu = load_frag(Us[cur], 40, wn * 32 + ni * 16, 0);
#pragma unroll
      for (int mi = 0; mi < 4; ++mi) {
        cg[mi][ni] = wmma_bf16(a[mi], bg, cg[mi][ni]);
        cu[mi][ni] = wmma_bf16(a[mi], bu, cu[mi][ni]);
      }
    }
    if (more) {
      cvt_store_w(greg, cur ? gDst0 : gDst1, true);
      cvt_store_w(ureg, cur ? uDst0 : uDst1, true);
      async_wait0();
    }
    __syncthreads();
  }

  const int mo = (lane >> 4) << 3;
  const int nn = lane & 15;
#pragma unroll
  for (int mi = 0; mi < 4; ++mi)
#pragma unroll
    for (int ni = 0; ni < 2; ++ni)
#pragma unroll
      for (int r = 0; r < 8; ++r) {
        const int mG = m0 + wm * 64 + mi * 16 + r + mo;
        const int n = n0 + wn * 32 + ni * 16 + nn;
        if (mG < cnt) {
          const float g = cg[mi][ni][r];
          const float u = cu[mi][ni][r];
          const float a = (g / (1.f + __expf(-g))) * u;  // silu(g)*u
          act[(size_t)(rowBase + mG) * Ih + n] = f2bf(a);
        }
      }
}

// ---------------------------------------------------------------------------
// Grouped expert down-projection with weighted scatter-add:
//   out[tok, n] += w_slot * (act_e[slot] * w2[e]^T)[n]     (atomic f32)
// ---------------------------------------------------------------------------
__global__ __launch_bounds__(256, 1) void gemm_down_scatter_kernel(
    const __bf16* __restrict__ act, int lda,
    const float* __restrict__ W2, long long wStrideE,
    const int* __restrict__ counts, const int* __restrict__ tok_list,
    const float* __restrict__ wt_list,
    float* __restrict__ out, int K, int N) {
  const int e = blockIdx.z;
  const int cnt = counts[e];
  const int m0 = blockIdx.x * 128;
  if (m0 >= cnt) return;
  const int n0 = blockIdx.y * 128;
  const float* W = W2 + (size_t)e * (size_t)wStrideE;
  const __bf16* Ae = act + (size_t)e * TOKENS * lda;

  __shared__ __bf16 As[2][128 * 40];
  __shared__ __bf16 Ws[2][128 * 40];
  const int tid = threadIdx.x;
  const int lane = tid & 31;
  const int wid = tid >> 5;
  const int wm = wid >> 2, wn = wid & 3;
  const int lr = tid >> 1;
  const int lc = (tid & 1) * 16;

  const int aRow = (m0 + lr < cnt) ? (m0 + lr) : 0;
  const __bf16* aBase = Ae + (size_t)aRow * lda + lc;
  const float* wBase = W + (size_t)(n0 + lr) * K + lc;  // N=2048, in range
  __bf16* aDst0 = &As[0][lr * 40 + lc];
  __bf16* aDst1 = &As[1][lr * 40 + lc];
  __bf16* wDst0 = &Ws[0][lr * 40 + lc];
  __bf16* wDst1 = &Ws[1][lr * 40 + lc];

  v8f c[4][2];
#pragma unroll
  for (int mi = 0; mi < 4; ++mi)
#pragma unroll
    for (int ni = 0; ni < 2; ++ni)
#pragma unroll
      for (int r = 0; r < 8; ++r) c[mi][ni][r] = 0.f;

  const int nk = K >> 5;
  float4 wreg[4];

  async_copy_b128(aBase, aDst0);
  async_copy_b128(aBase + 8, aDst0 + 8);
  {
    const float4* s = (const float4*)wBase;
    wreg[0] = s[0]; wreg[1] = s[1]; wreg[2] = s[2]; wreg[3] = s[3];
  }
  cvt_store_w(wreg, wDst0, true);
  async_wait0();
  __syncthreads();

  for (int kk = 0; kk < nk; ++kk) {
    const int cur = kk & 1;
    const bool more = (kk + 1) < nk;
    if (more) {
      const __bf16* an = aBase + (size_t)(kk + 1) * 32;
      async_copy_b128(an, cur ? aDst0 : aDst1);
      async_copy_b128(an + 8, (cur ? aDst0 : aDst1) + 8);
      const float4* s = (const float4*)(wBase + (size_t)(kk + 1) * 32);
      wreg[0] = s[0]; wreg[1] = s[1]; wreg[2] = s[2]; wreg[3] = s[3];
    }
    v16bf a[4];
#pragma unroll
    for (int mi = 0; mi < 4; ++mi)
      a[mi] = load_frag(As[cur], 40, wm * 64 + mi * 16, 0);
#pragma unroll
    for (int ni = 0; ni < 2; ++ni) {
      v16bf b = load_frag(Ws[cur], 40, wn * 32 + ni * 16, 0);
#pragma unroll
      for (int mi = 0; mi < 4; ++mi)
        c[mi][ni] = wmma_bf16(a[mi], b, c[mi][ni]);
    }
    if (more) {
      cvt_store_w(wreg, cur ? wDst0 : wDst1, true);
      async_wait0();
    }
    __syncthreads();
  }

  const int mo = (lane >> 4) << 3;
  const int nn = lane & 15;
#pragma unroll
  for (int mi = 0; mi < 4; ++mi)
#pragma unroll
    for (int ni = 0; ni < 2; ++ni)
#pragma unroll
      for (int r = 0; r < 8; ++r) {
        const int mG = m0 + wm * 64 + mi * 16 + r + mo;
        const int n = n0 + wn * 32 + ni * 16 + nn;
        if (mG < cnt) {
          const int t = tok_list[e * TOKENS + mG];
          const float wsl = wt_list[e * TOKENS + mG];
          atomicAdd(out + (size_t)t * N + n, wsl * c[mi][ni][r]);
        }
      }
}

// ---------------------------------------------------------------------------
// Flash attention (causal), WMMA for QK^T and PV.
// Block = 256 thr = 8 waves; 128 q-rows/block (16 per wave); 64-key tiles.
// q: bf16 [T, NH*QHD] (first NOPE cols/head); kv: bf16 [T, 4160].
// Q/K tiles staged via async global->LDS; V transposed through VGPRs.
// ---------------------------------------------------------------------------
__global__ __launch_bounds__(256, 1) void attn_kernel(
    const __bf16* __restrict__ q, const __bf16* __restrict__ kv,
    __bf16* __restrict__ attn_out) {
  const int qt = blockIdx.x;
  const int head = blockIdx.y;
  const int b = blockIdx.z;
  const int q0 = qt * 128;
  const int tid = threadIdx.x;
  const int lane = tid & 31;
  const int w = tid >> 5;

  __shared__ __bf16 Qs[128 * 136];
  __shared__ __bf16 Ks[64 * 136];
  __shared__ __bf16 Vt[128 * 72];       // V transposed: [d][k]
  __shared__ __bf16 Ps[8 * 16 * 72];    // per-wave P staging

  for (int i = tid; i < 128 * 16; i += 256) {
    const int r = i >> 4, cc = (i & 15) * 8;
    async_copy_b128(q + (size_t)(b * 1024 + q0 + r) * (NHEAD * QHD) +
                        head * QHD + cc,
                    Qs + r * 136 + cc);
  }

  v8f o[8];
#pragma unroll
  for (int di = 0; di < 8; ++di)
#pragma unroll
    for (int r = 0; r < 8; ++r) o[di][r] = 0.f;
  float m_run[8], l_run[8];
#pragma unroll
  for (int r = 0; r < 8; ++r) { m_run[r] = -1e30f; l_run[r] = 0.f; }

  const int mo = (lane >> 4) << 3;
  const int nn = lane & 15;
  const float scale = 0.07216878364870323f;  // QHD^-0.5 = 192^-0.5
  const int nkt = (q0 + 128) / 64;

  for (int kt = 0; kt < nkt; ++kt) {
    __syncthreads();   // Ks/Vt reuse fence
    for (int i = tid; i < 64 * 16; i += 256) {
      const int r = i >> 4, cc = (i & 15) * 8;
      async_copy_b128(kv + (size_t)(b * 1024 + kt * 64 + r) * 4160 +
                          head * NOPE + cc,
                      Ks + r * 136 + cc);
    }
    for (int i = tid; i < 64 * 16; i += 256) {
      const int r = i >> 4, cc = (i & 15) * 8;
      union { uint4 u; __bf16 h[8]; } tv;
      tv.u = *(const uint4*)(kv + (size_t)(b * 1024 + kt * 64 + r) * 4160 +
                             NHEAD * NOPE + head * VDIM + cc);
#pragma unroll
      for (int j = 0; j < 8; ++j) Vt[(cc + j) * 72 + r] = tv.h[j];
    }
    async_wait0();     // Q (first iter) + K copies visible in LDS
    __syncthreads();

    v8f sc[4];
#pragma unroll
    for (int ni = 0; ni < 4; ++ni)
#pragma unroll
      for (int r = 0; r < 8; ++r) sc[ni][r] = 0.f;

#pragma unroll
    for (int kk = 0; kk < 4; ++kk) {
      v16bf a = load_frag(Qs, 136, w * 16, kk * 32);
#pragma unroll
      for (int ni = 0; ni < 4; ++ni) {
        v16bf bq = load_frag(Ks, 136, ni * 16, kk * 32);
        sc[ni] = wmma_bf16(a, bq, sc[ni]);
      }
    }

    __bf16* Pw = Ps + w * 16 * 72;
#pragma unroll
    for (int r = 0; r < 8; ++r) {
      const int rowg = q0 + w * 16 + r + mo;
      float mloc = -1e30f;
#pragma unroll
      for (int ni = 0; ni < 4; ++ni) {
        const int colg = kt * 64 + ni * 16 + nn;
        float v = sc[ni][r] * scale;
        if (colg > rowg) v = -1e30f;    // causal mask
        sc[ni][r] = v;
        mloc = fmaxf(mloc, v);
      }
#pragma unroll
      for (int off = 8; off >= 1; off >>= 1)
        mloc = fmaxf(mloc, __shfl_xor(mloc, off, 16));
      const float mnew = fmaxf(m_run[r], mloc);
      const float alpha = __expf(m_run[r] - mnew);
      float rs = 0.f;
#pragma unroll
      for (int ni = 0; ni < 4; ++ni) {
        const float p = __expf(sc[ni][r] - mnew);
        Pw[(r + mo) * 72 + ni * 16 + nn] = f2bf(p);
        rs += p;
      }
#pragma unroll
      for (int off = 8; off >= 1; off >>= 1) rs += __shfl_xor(rs, off, 16);
      l_run[r] = l_run[r] * alpha + rs;
      m_run[r] = mnew;
#pragma unroll
      for (int di = 0; di < 8; ++di) o[di][r] *= alpha;
    }

    // PV: same-wave LDS RAW on Pw (LDS is in-order per wave)
#pragma unroll
    for (int kk2 = 0; kk2 < 2; ++kk2) {
      v16bf a = load_frag(Pw, 72, 0, kk2 * 32);
#pragma unroll
      for (int di = 0; di < 8; ++di) {
        v16bf bv = load_frag(Vt, 72, di * 16, kk2 * 32);
        o[di] = wmma_bf16(a, bv, o[di]);
      }
    }
  }

#pragma unroll
  for (int di = 0; di < 8; ++di)
#pragma unroll
    for (int r = 0; r < 8; ++r) {
      const int rowg = b * 1024 + q0 + w * 16 + r + mo;
      attn_out[(size_t)rowg * (NHEAD * VDIM) + head * VDIM + di * 16 + nn] =
          f2bf(o[di][r] / l_run[r]);
    }
}

// ---------------------------------------------------------------------------
// Router: logits = x2 @ gate_w^T, top-4 softmax, build per-expert token lists.
// ---------------------------------------------------------------------------
__global__ void zero_counts_kernel(int* c) {
  if (threadIdx.x < NEXP) c[threadIdx.x] = 0;
}

__global__ __launch_bounds__(256) void gate_topk_kernel(
    const __bf16* __restrict__ x2, const float* __restrict__ gw,
    int* __restrict__ counts, int* __restrict__ tok_list,
    float* __restrict__ wt_list) {
  const int t = blockIdx.x;
  __shared__ float logits[NEXP];
  const int ge = threadIdx.x >> 4;   // expert 0..15
  const int l16 = threadIdx.x & 15;
  float p = 0.f;
  const __bf16* xr = x2 + (size_t)t * HDIM;
  const float* wr = gw + (size_t)ge * HDIM;
  for (int k = l16; k < HDIM; k += 16) p += bf2f(xr[k]) * wr[k];
#pragma unroll
  for (int off = 8; off >= 1; off >>= 1) p += __shfl_xor(p, off, 16);
  if (l16 == 0) logits[ge] = p;
  __syncthreads();
  if (threadIdx.x == 0) {
    unsigned used = 0;
    float sv[4]; int si[4];
    for (int j = 0; j < 4; ++j) {
      float bv = -1e30f; int bi = 0;
      for (int i = 0; i < NEXP; ++i)
        if (!((used >> i) & 1u) && logits[i] > bv) { bv = logits[i]; bi = i; }
      used |= 1u << bi; sv[j] = bv; si[j] = bi;
    }
    // top-k of softmax then renormalize == softmax over the top-k logits
    const float mx = sv[0];
    float s = 0.f;
    for (int j = 0; j < 4; ++j) { sv[j] = __expf(sv[j] - mx); s += sv[j]; }
    for (int j = 0; j < 4; ++j) {
      const int e = si[j];
      const int pos = atomicAdd(counts + e, 1);
      tok_list[e * TOKENS + pos] = t;
      wt_list[e * TOKENS + pos] = sv[j] / s;
    }
  }
}

// ---------------------------------------------------------------------------
// Host-side orchestration
// ---------------------------------------------------------------------------
extern "C" void kernel_launch(void* const* d_in, const int* in_sizes, int n_in,
                              void* d_out, int out_size, void* d_ws,
                              size_t ws_size, hipStream_t stream) {
  (void)in_sizes; (void)n_in; (void)out_size; (void)ws_size;
  const float* hidden    = (const float*)d_in[0];
  /* d_in[1] = positions: unused (reference never applies RoPE) */
  const float* ln1_w     = (const float*)d_in[2];
  const float* ln2_w     = (const float*)d_in[3];
  const float* q_w       = (const float*)d_in[4];   // [3072, 2048]
  const float* kv_w      = (const float*)d_in[5];   // [4160, 2048]
  const float* o_w       = (const float*)d_in[6];   // [2048, 2048]
  const float* gate_w    = (const float*)d_in[7];   // [16, 2048]
  const float* w1        = (const float*)d_in[8];   // [16, 2816, 2048]
  const float* w2        = (const float*)d_in[9];   // [16, 2048, 1408]
  const float* sh_gu_w   = (const float*)d_in[10];  // [5632, 2048]
  const float* sh_down_w = (const float*)d_in[11];  // [2048, 2816]
  float* out = (float*)d_out;

  char* wp = (char*)d_ws;
  auto alloc = [&](size_t bytes) -> void* {
    void* p = (void*)wp;
    wp += (bytes + 255) & ~(size_t)255;
    return p;
  };
  __bf16* x_bf    = (__bf16*)alloc((size_t)TOKENS * HDIM * 2);
  __bf16* q_bf    = (__bf16*)alloc((size_t)TOKENS * 3072 * 2);
  __bf16* kv_bf   = (__bf16*)alloc((size_t)TOKENS * 4160 * 2);
  __bf16* attn_bf = (__bf16*)alloc((size_t)TOKENS * HDIM * 2);
  float*  hid     = (float*)alloc((size_t)TOKENS * HDIM * 4);
  __bf16* x2_bf   = (__bf16*)alloc((size_t)TOKENS * HDIM * 2);
  __bf16* act_sh  = (__bf16*)alloc((size_t)TOKENS * SHIDIM * 2);
  int*    counts  = (int*)alloc(NEXP * 4);
  int*    tok_lst = (int*)alloc((size_t)NEXP * TOKENS * 4);
  float*  wt_lst  = (float*)alloc((size_t)NEXP * TOKENS * 4);
  __bf16* act_e   = (__bf16*)alloc((size_t)NEXP * TOKENS * IDIM * 2);

  const dim3 blk(256);

  // 1) x = rmsnorm(hidden, ln1) -> bf16
  rmsnorm_kernel<<<TOKENS, blk, 0, stream>>>(hidden, ln1_w, x_bf, HDIM);
  // 2) q = x @ q_w^T            (bf16 out)
  gemm_dense_kernel<<<dim3(16, 24), blk, 0, stream>>>(
      x_bf, HDIM, q_w, HDIM, nullptr, q_bf, nullptr, TOKENS, 3072, HDIM);
  // 3) kv = x @ kv_w^T          (bf16 out, partial last N tile)
  gemm_dense_kernel<<<dim3(16, 33), blk, 0, stream>>>(
      x_bf, HDIM, kv_w, HDIM, nullptr, kv_bf, nullptr, TOKENS, 4160, HDIM);
  // 4) attention -> bf16 [T, 2048]
  attn_kernel<<<dim3(8, NHEAD, 2), blk, 0, stream>>>(q_bf, kv_bf, attn_bf);
  // 5) hid = hidden + attn @ o_w^T
  gemm_dense_kernel<<<dim3(16, 16), blk, 0, stream>>>(
      attn_bf, HDIM, o_w, HDIM, hid, nullptr, hidden, TOKENS, HDIM, HDIM);
  // 6) x2 = rmsnorm(hid, ln2) -> bf16
  rmsnorm_kernel<<<TOKENS, blk, 0, stream>>>(hid, ln2_w, x2_bf, HDIM);
  // 7) routing: top-4 experts per token
  zero_counts_kernel<<<1, 32, 0, stream>>>(counts);
  gate_topk_kernel<<<TOKENS, blk, 0, stream>>>(x2_bf, gate_w, counts,
                                               tok_lst, wt_lst);
  // 8) shared expert gate/up + SiLU (dual-GEMM fused)
  gemm_gu_silu_kernel<<<dim3(16, 22, 1), blk, 0, stream>>>(
      x2_bf, HDIM, sh_gu_w, 0LL, SHIDIM, HDIM, nullptr, nullptr, TOKENS,
      act_sh);
  // 9) out = hid + act_sh @ sh_down_w^T
  gemm_dense_kernel<<<dim3(16, 16), blk, 0, stream>>>(
      act_sh, SHIDIM, sh_down_w, SHIDIM, out, nullptr, hid, TOKENS, HDIM,
      SHIDIM);
  // 10) routed experts: gathered gate/up + SiLU (grouped; blocks early-exit)
  gemm_gu_silu_kernel<<<dim3(16, 11, NEXP), blk, 0, stream>>>(
      x2_bf, HDIM, w1, (long long)2 * IDIM * HDIM, IDIM, HDIM, counts,
      tok_lst, TOKENS, act_e);
  // 11) expert down-proj, weighted atomic scatter-add into out
  gemm_down_scatter_kernel<<<dim3(16, 16, NEXP), blk, 0, stream>>>(
      act_e, IDIM, w2, (long long)HDIM * IDIM, counts, tok_lst, wt_lst, out,
      IDIM, HDIM);
}
